// GraphHighway_88751204205025
// MI455X (gfx1250) — matrix operand order repaired and checked by reading, not compiled, not run on hardware
//
#include <hip/hip_runtime.h>

typedef __attribute__((ext_vector_type(8)))  float   v8f;
typedef __attribute__((ext_vector_type(8)))  __bf16  v8bf;
typedef __attribute__((ext_vector_type(16))) __bf16  v16bf;

#define N_NODES  20000
#define DIM      256
#define N_REL    5
#define N_NEIGH  10
#define KDIM     (DIM * (1 + N_REL))   // 1536 = [x | context_0..4]
#define NDIM     (2 * DIM)             // 512  = [h cols | gate cols]
#define KT_COUNT (KDIM / 32)           // 48 k-steps of 32
#define STAGES   6                     // 6 stages x 8 k-steps
#define STEPS_PER_STAGE 8

__device__ __forceinline__ __bf16 f2bf(float f) {
  unsigned u = __builtin_bit_cast(unsigned, f);
  u += 0x7FFFu + ((u >> 16) & 1u);          // round-to-nearest-even
  unsigned short h = (unsigned short)(u >> 16);
  return __builtin_bit_cast(__bf16, h);
}

__device__ __forceinline__ v16bf cat16(v8bf lo, v8bf hi) {
  return __builtin_shufflevector(lo, hi, 0,1,2,3,4,5,6,7,8,9,10,11,12,13,14,15);
}

// ---------------------------------------------------------------------------
// Kernel 1: A = [x | masked-mean context] as bf16, (N_NODES x KDIM).
// One block per node, one thread per channel. Gathered neighbor rows are 1KB
// coalesced reads that hit L2 (x = 20MB << 192MB L2).
// ---------------------------------------------------------------------------
__global__ void build_A(const float* __restrict__ x,
                        const long long* __restrict__ rel,
                        const float* __restrict__ rel_mask,
                        __bf16* __restrict__ Abf) {
  const int n = blockIdx.x;
  const int d = threadIdx.x;
  __bf16* row = Abf + (size_t)n * KDIM;
  row[d] = f2bf(x[(size_t)n * DIM + d]);

  const long long* reln = rel + (size_t)n * N_REL * N_NEIGH;
  const float* mmul = rel_mask + (size_t)n * 2 * N_REL * N_NEIGH; // [n][0][r][j]
  const float* mdiv = mmul + N_REL * N_NEIGH;                     // [n][1][r][j]
  for (int r = 0; r < N_REL; ++r) {
    float acc = 0.f, div = 0.f;
    #pragma unroll
    for (int j = 0; j < N_NEIGH; ++j) {
      long long idx = reln[r * N_NEIGH + j];
      acc += x[(size_t)idx * DIM + d] * mmul[r * N_NEIGH + j];
      div += mdiv[r * N_NEIGH + j];
    }
    row[DIM + r * DIM + d] = f2bf(acc / div);
  }
}

// ---------------------------------------------------------------------------
// Kernel 2: pack B = [[W,W_carry],[V,V_carry]] (KDIM x NDIM) as bf16 in WMMA
// 32x16 B-fragment order: frag(nt,kt) = 32 lanes x 16 bf16 contiguous.
// lanes 0-15: col nt*16+lane, K = kt*32+{0..7,16..23}; lanes 16-31: K = kt*32+{8..15,24..31}.
// ---------------------------------------------------------------------------
__global__ void pack_B(const float* __restrict__ W,  const float* __restrict__ Wc,
                       const float* __restrict__ V,  const float* __restrict__ Vc,
                       __bf16* __restrict__ Bf) {
  int idx = blockIdx.x * blockDim.x + threadIdx.x;   // flat fragment-order index
  if (idx >= KDIM * NDIM) return;
  int i    = idx & 15;
  int lane = (idx >> 4) & 31;
  int rest = idx >> 9;
  int kt   = rest % KT_COUNT;
  int nt   = rest / KT_COUNT;

  int n = nt * 16 + (lane & 15);                  // output column (0..511)
  int k = kt * 32 + ((lane >> 4) << 3) + (i & 7) + ((i >> 3) << 4);       // K row (0..1535)
  int e = n & (DIM - 1);
  bool gate = (n >= DIM);
  float v;
  if (k < DIM) {
    v = gate ? Wc[k * DIM + e] : W[k * DIM + e];
  } else {
    int kk = k - DIM;                             // r*256 + d
    v = gate ? Vc[(size_t)kk * DIM + e] : V[(size_t)kk * DIM + e];
  }
  Bf[idx] = f2bf(v);
}

// ---------------------------------------------------------------------------
// Kernel 3: bf16 WMMA GEMM + highway epilogue with async LDS staging of A.
//  - block = 8 waves, owns 32 A-rows; wave w owns h-cols [32w,32w+32) and the
//    matching gate cols (+256) -> epilogue is wave-local.
//  - K processed in 6 stages of 8 k-steps; each stage's 16KB A slab is
//    async-copied to LDS (global_load_async_to_lds_b128, double buffered,
//    s_wait_asynccnt) and consumed via ds_load_b128.
//  - 8 v_wmma_f32_16x16x32_bf16 per k-step per wave.
// ---------------------------------------------------------------------------
__global__ void __launch_bounds__(256)
gemm_highway(const __bf16* __restrict__ Abf, const __bf16* __restrict__ Bf,
             const float* __restrict__ x, const float* __restrict__ b,
             const float* __restrict__ bcar, float* __restrict__ out) {
  __shared__ __align__(16) __bf16 smem[2][32][8 * 32];   // 2 x 16KB

  const int tid  = threadIdx.x;
  const int lane = tid & 31;
  const int colg = tid >> 5;                 // wave id = h column group (0..7)
  const int m0   = blockIdx.x * 32;

  const __bf16* Arow = Abf + (size_t)m0 * KDIM;

  // Issue one stage's cooperative async copy: 32 rows x 512B, 16B per lane-op.
  auto copy_stage = [&](int st, int bufIdx) {
    const int k0 = st * (STEPS_PER_STAGE * 32);
    #pragma unroll
    for (int it = 0; it < 4; ++it) {
      int c   = it * 256 + tid;              // chunk 0..1023
      int row = c >> 5;
      int cib = c & 31;                      // 16B chunk within row
      const __bf16* g = Arow + (size_t)row * KDIM + k0 + cib * 8;
      unsigned lds = (unsigned)(size_t)&smem[bufIdx][row][cib * 8];
      asm volatile("global_load_async_to_lds_b128 %0, %1, off"
                   :: "v"(lds), "v"(g) : "memory");
    }
  };

  // B fragment base pointers (pre-packed lane-major, 1KB per 32x16 fragment).
  const __bf16* bh_base0 = Bf + ((size_t)(colg * 2 + 0)      * KT_COUNT * 32 + lane) * 16;
  const __bf16* bh_base1 = Bf + ((size_t)(colg * 2 + 1)      * KT_COUNT * 32 + lane) * 16;
  const __bf16* bg_base0 = Bf + ((size_t)(16 + colg * 2 + 0) * KT_COUNT * 32 + lane) * 16;
  const __bf16* bg_base1 = Bf + ((size_t)(16 + colg * 2 + 1) * KT_COUNT * 32 + lane) * 16;

  v8f cH[2][2] = {};   // [row half][col frag]
  v8f cG[2][2] = {};

  copy_stage(0, 0);
  asm volatile("s_wait_asynccnt 0x0" ::: "memory");
  __syncthreads();

  for (int st = 0; st < STAGES; ++st) {
    const int cur = st & 1;
    if (st + 1 < STAGES) copy_stage(st + 1, cur ^ 1);   // overlap next slab

    #pragma unroll
    for (int s = 0; s < STEPS_PER_STAGE; ++s) {
      const int kt = st * STEPS_PER_STAGE + s;

      // A fragments from LDS (ISA 16-bit A layout):
      // lanes 0-15: M=lane, K={0..7,16..23}; lanes 16-31: M=lane-16, K={8..15,24..31}
      const __bf16* ar0 = &smem[cur][lane & 15][s * 32 + ((lane < 16) ? 0 : 8)];
      v16bf a0 = cat16(*(const v8bf*)(ar0), *(const v8bf*)(ar0 + 16));
      const __bf16* ar1 = ar0 + 16 * (8 * 32);
      v16bf a1 = cat16(*(const v8bf*)(ar1), *(const v8bf*)(ar1 + 16));

      // B fragments from L2 (this lane's 16 bf16 are contiguous).
      const size_t ko = (size_t)kt * (32 * 16);
      v16bf bh0 = cat16(*(const v8bf*)(bh_base0 + ko), *(const v8bf*)(bh_base0 + ko + 8));
      v16bf bh1 = cat16(*(const v8bf*)(bh_base1 + ko), *(const v8bf*)(bh_base1 + ko + 8));
      v16bf bg0 = cat16(*(const v8bf*)(bg_base0 + ko), *(const v8bf*)(bg_base0 + ko + 8));
      v16bf bg1 = cat16(*(const v8bf*)(bg_base1 + ko), *(const v8bf*)(bg_base1 + ko + 8));

      cH[0][0] = __builtin_amdgcn_wmma_f32_16x16x32_bf16(false, a0, false, bh0, (short)0, cH[0][0], false, false);
      cH[1][0] = __builtin_amdgcn_wmma_f32_16x16x32_bf16(false, a1, false, bh0, (short)0, cH[1][0], false, false);
      cH[0][1] = __builtin_amdgcn_wmma_f32_16x16x32_bf16(false, a0, false, bh1, (short)0, cH[0][1], false, false);
      cH[1][1] = __builtin_amdgcn_wmma_f32_16x16x32_bf16(false, a1, false, bh1, (short)0, cH[1][1], false, false);
      cG[0][0] = __builtin_amdgcn_wmma_f32_16x16x32_bf16(false, a0, false, bg0, (short)0, cG[0][0], false, false);
      cG[1][0] = __builtin_amdgcn_wmma_f32_16x16x32_bf16(false, a1, false, bg0, (short)0, cG[1][0], false, false);
      cG[0][1] = __builtin_amdgcn_wmma_f32_16x16x32_bf16(false, a0, false, bg1, (short)0, cG[0][1], false, false);
      cG[1][1] = __builtin_amdgcn_wmma_f32_16x16x32_bf16(false, a1, false, bg1, (short)0, cG[1][1], false, false);
    }

    asm volatile("s_wait_asynccnt 0x0" ::: "memory");  // next slab landed
    __syncthreads();                                    // all waves done with cur
  }

  // Epilogue: C/D layout -> VGPR v holds row m0 + v (+8 for lanes 16-31).
  const int eA = colg * 32 + (lane & 15);
  const int eB = eA + 16;
  const float bvA  = b[eA],    bvB  = b[eB];
  const float bcvA = bcar[eA], bcvB = bcar[eB];
  const int mlane = m0 + ((lane < 16) ? 0 : 8);

  #pragma unroll
  for (int v = 0; v < 8; ++v) {
    #pragma unroll
    for (int rh = 0; rh < 2; ++rh) {
      const int r = mlane + rh * 16 + v;
      const size_t ro = (size_t)r * DIM;
      float h0 = cH[rh][0][v] + bvA;
      float g0 = 1.f / (1.f + __expf(-(cG[rh][0][v] + bcvA)));
      out[ro + eA] = g0 * h0 + (1.f - g0) * x[ro + eA];
      float h1 = cH[rh][1][v] + bvB;
      float g1 = 1.f / (1.f + __expf(-(cG[rh][1][v] + bcvB)));
      out[ro + eB] = g1 * h1 + (1.f - g1) * x[ro + eB];
    }
  }
}

// ---------------------------------------------------------------------------
extern "C" void kernel_launch(void* const* d_in, const int* in_sizes, int n_in,
                              void* d_out, int out_size, void* d_ws, size_t ws_size,
                              hipStream_t stream) {
  const float*     x        = (const float*)d_in[0];
  const long long* rel      = (const long long*)d_in[1];
  const float*     rel_mask = (const float*)d_in[2];
  const float*     W        = (const float*)d_in[3];
  const float*     Wc       = (const float*)d_in[4];
  const float*     V        = (const float*)d_in[5];
  const float*     Vc       = (const float*)d_in[6];
  const float*     b        = (const float*)d_in[7];
  const float*     bcar     = (const float*)d_in[8];
  float* out = (float*)d_out;

  __bf16* Abf = (__bf16*)d_ws;                       // 20000*1536 bf16 = 61.44 MB
  __bf16* Bf  = Abf + (size_t)N_NODES * KDIM;        // 1536*512  bf16 =  1.57 MB

  build_A<<<N_NODES, DIM, 0, stream>>>(x, rel, rel_mask, Abf);
  pack_B<<<(KDIM * NDIM + 255) / 256, 256, 0, stream>>>(W, Wc, V, Vc, Bf);
  gemm_highway<<<N_NODES / 32, 256, 0, stream>>>(Abf, Bf, x, b, bcar, out);
}
